// SoftVectorQuantizer_87041807220993
// MI455X (gfx1250) — matrix operand order repaired and checked by reading, not compiled, not run on hardware
//
#include <hip/hip_runtime.h>

// MI455X (gfx1250) Soft Vector Quantizer, fused, bf16-WMMA with f32 accumulate.
// B=65536, K=1024, D=256.  out = [quantized (B*D f32), loss (1 f32)]
//
// E (1MB, L2-resident) is pre-converted ONCE into two bf16 buffers laid out in
// exact WMMA B-fragment order, so every B operand in the hot loops is two
// contiguous global_load_b128 per lane (coalesced 1KB/wave), no cvt, no LDS
// staging, no transpose.

typedef __attribute__((ext_vector_type(16))) __bf16 v16bf;
typedef __attribute__((ext_vector_type(8)))  __bf16 v8bf;
typedef __attribute__((ext_vector_type(8)))  float  v8f;
typedef __attribute__((ext_vector_type(4)))  float  v4f;

constexpr int Bn = 65536, Kn = 1024, Dn = 256;
constexpr int ROWS = 16;            // x rows per block
constexpr int NBLK = Bn / ROWS;     // 4096 blocks
constexpr int NTHR = 128;           // 4 waves (wave32)

// padded LDS strides (elements) to dodge bank conflicts
constexpr int LSTR  = Kn + 4;       // logits f32 row stride (1028)
constexpr int XBSTR = Dn + 8;       // x bf16 row stride (264)
constexpr int PSTR  = Kn + 8;       // P bf16 row stride (1032)

// LDS layout (bytes)
constexpr int OFF_LG   = 0;                           // 16*1028*4 = 65792
constexpr int OFF_XB   = 65792;                       // 16*264*2  = 8448
constexpr int OFF_P    = OFF_XB + ROWS * XBSTR * 2;   // 16*1032*2 = 33024
constexpr int OFF_XX   = OFF_P + ROWS * PSTR * 2;
constexpr int OFF_PART = OFF_XX + 64;                 // 4 waves * 16 rows * 4B
constexpr int OFF_RMAX = OFF_PART + 256;
constexpr int OFF_RINV = OFF_RMAX + 64;
constexpr int OFF_WL   = OFF_RINV + 64;
constexpr int SMEM_SZ  = OFF_WL + 64;

// workspace layout (bytes)
constexpr size_t WS_EE    = 0;                       // Kn f32
constexpr size_t WS_BPART = 4096;                    // NBLK f32
constexpr size_t WS_EW1   = 20480;                   // 64kt*8d8*512 bf16 = 512KB
constexpr size_t WS_EW2   = 544768;                  // 16dt*32kc*512 bf16 = 512KB

#define SHUF16(a, b) __builtin_shufflevector(a, b, 0,1,2,3,4,5,6,7,8,9,10,11,12,13,14,15)

// ---------------- prep 0: codebook row norms ----------------
__global__ void svq_prep_ee(const float* __restrict__ E, float* __restrict__ ee) {
    __shared__ float red[256];
    const int k = blockIdx.x, t = threadIdx.x;
    const float v = E[(size_t)k * Dn + t];
    red[t] = v * v;
    __syncthreads();
    for (int o = 128; o > 0; o >>= 1) {
        if (t < o) red[t] += red[t + o];
        __syncthreads();
    }
    if (t == 0) ee[k] = red[0];
}

// ---------------- prep 1: E -> bf16 fragments for GEMM1 (B = E^T, 32x16) ----
// packet(kt, d8, lane, h): 8 bf16 = E[kt*16 + (lane&15), d8*32 + 8*(lane>>4) + 16h + 0..7]
__global__ void svq_prep_w1(const float* __restrict__ E, __bf16* __restrict__ Ew1) {
    const int b = blockIdx.x;            // kt*8 + d8   (512 blocks)
    const int kt = b >> 3, d8 = b & 7;
    const int t = threadIdx.x;           // 64 threads: lane*2 + h
    const int lane = t >> 1, h = t & 1;
    const int hl = lane & 15, g = lane >> 4;
    const float* src = E + (size_t)(kt * 16 + hl) * Dn + d8 * 32 + 8 * g + 16 * h;
    __bf16* dst = Ew1 + ((size_t)b * 64 + t) * 8;
#pragma unroll
    for (int i = 0; i < 8; i++) dst[i] = (__bf16)src[i];
}

// ---------------- prep 2: E -> bf16 fragments for GEMM2 (B = E, 32x16) ------
// packet(dt, kc, lane, h): 8 bf16 = E[kc*32 + 8*(lane>>4) + 16h + i, dt*16 + (lane&15)]
__global__ void svq_prep_w2(const float* __restrict__ E, __bf16* __restrict__ Ew2) {
    const int b = blockIdx.x;            // dt*32 + kc  (512 blocks)
    const int dt = b >> 5, kc = b & 31;
    const int t = threadIdx.x;           // 64 threads: lane*2 + h
    const int lane = t >> 1, h = t & 1;
    const int hl = lane & 15, g = lane >> 4;
    const float* src = E + (size_t)(kc * 32 + 8 * g + 16 * h) * Dn + dt * 16 + hl;
    __bf16* dst = Ew2 + ((size_t)b * 64 + t) * 8;
#pragma unroll
    for (int i = 0; i < 8; i++) dst[i] = (__bf16)src[(size_t)i * Dn];
}

// ---------------- main: fused SVQ ----------------
__global__ __launch_bounds__(NTHR) void svq_main(
    const float* __restrict__ x, const float* __restrict__ var,
    const __bf16* __restrict__ Ew1, const __bf16* __restrict__ Ew2,
    const float* __restrict__ ee,
    float* __restrict__ out, float* __restrict__ bpart) {

    extern __shared__ char smem[];
    float*  lg   = (float*)(smem + OFF_LG);    // logits [16][LSTR] f32
    __bf16* xb   = (__bf16*)(smem + OFF_XB);   // x tile bf16 [16][XBSTR]
    __bf16* Pm   = (__bf16*)(smem + OFF_P);    // probs bf16 [16][PSTR]
    float*  xxs  = (float*)(smem + OFF_XX);    // row |x|^2
    float*  part = (float*)(smem + OFF_PART);  // per-wave partials [4][16]
    float*  rmx  = (float*)(smem + OFF_RMAX);
    float*  rinv = (float*)(smem + OFF_RINV);
    float*  wl   = (float*)(smem + OFF_WL);

    const int t = threadIdx.x;
    const int w = t >> 5, lane = t & 31, g = lane >> 4, hl = lane & 15;
    const int row0 = blockIdx.x * ROWS;

    // ---- Phase 0: stage x tile (bf16) + per-row sum of squares ----
    if (t < 16) xxs[t] = 0.f;
    __syncthreads();
    {
        const int row = t >> 3, d0 = (t & 7) * 32;
        const v4f* xp = (const v4f*)(x + (size_t)(row0 + row) * Dn + d0);
        float ss = 0.f;
#pragma unroll
        for (int j = 0; j < 8; j++) {
            v4f v = xp[j];
#pragma unroll
            for (int ii = 0; ii < 4; ii++) {
                xb[row * XBSTR + d0 + 4 * j + ii] = (__bf16)v[ii];
                ss += v[ii] * v[ii];
            }
        }
        atomicAdd(&xxs[row], ss);
    }
    __syncthreads();

    // ---- Phase 1: logits = var - 0.5*dist*exp(-2var), via WMMA (x @ E^T) ----
    // hoist the k-invariant A fragments (x tile) into registers: 8 x v16bf
    v16bf Af[8];
#pragma unroll
    for (int d8 = 0; d8 < 8; d8++) {
        const __bf16* ap = xb + hl * XBSTR + d8 * 32 + 8 * g;
        v8bf alo = *(const v8bf*)ap;
        v8bf ahi = *(const v8bf*)(ap + 16);
        Af[d8] = SHUF16(alo, ahi);
    }

    float rmax[8];
#pragma unroll
    for (int r = 0; r < 8; r++) rmax[r] = -3.0e38f;

    const __bf16* ew1l = Ew1 + (size_t)lane * 16;
    for (int it = 0; it < 16; it++) {
        const int kt = w * 16 + it;          // this wave owns k-tiles [16w, 16w+16)
        const int k0 = kt * 16;
        const int kg = k0 + hl;
        const float eek = ee[kg];

        v8f acc;
#pragma unroll
        for (int i = 0; i < 8; i++) acc[i] = 0.f;

#pragma unroll
        for (int d8 = 0; d8 < 8; d8++) {
            const __bf16* bp = ew1l + (size_t)(kt * 8 + d8) * 512;
            v8bf b0 = *(const v8bf*)bp;
            v8bf b1 = *(const v8bf*)(bp + 8);
            v16bf Bf = SHUF16(b0, b1);
            acc = __builtin_amdgcn_wmma_f32_16x16x32_bf16(
                false, Af[d8], false, Bf, (short)0, acc, false, false);
        }
        // epilogue: C layout m = r + 8g, n = hl
        const float* vp = var + (size_t)row0 * Kn + kg;
#pragma unroll
        for (int r = 0; r < 8; r++) {
            const int m = r + 8 * g;
            const float vv = vp[(size_t)m * Kn];
            const float s = __expf(-2.f * vv);
            const float dist = xxs[m] + eek - 2.f * acc[r];
            const float logit = vv - 0.5f * dist * s;   // log prob (unnormalized)
            lg[m * LSTR + k0 + hl] = logit;
            rmax[r] = fmaxf(rmax[r], logit);
        }
    }
    // reduce row max across the 16 lanes sharing a row, then across waves
#pragma unroll
    for (int r = 0; r < 8; r++) {
        float v = rmax[r];
        for (int o = 1; o < 16; o <<= 1) v = fmaxf(v, __shfl_xor(v, o, 32));
        if (hl == 0) part[w * 16 + r + 8 * g] = v;
    }
    __syncthreads();
    if (t < 16) {
        float mx = part[t];
        for (int ww = 1; ww < 4; ww++) mx = fmaxf(mx, part[ww * 16 + t]);
        rmx[t] = mx;
    }
    __syncthreads();

    // ---- Phase 2: P = exp(logit - rowmax) as bf16, plus row sums ----
    {
        const int kb = 256 * w;
        for (int m = 0; m < 16; m++) {
            const float rm = rmx[m];
            float sacc = 0.f;
            for (int kk = lane; kk < 256; kk += 32) {
                const int k = kb + kk;
                const float p = __expf(lg[m * LSTR + k] - rm);
                Pm[m * PSTR + k] = (__bf16)p;
                sacc += p;
            }
            for (int o = 1; o < 32; o <<= 1) sacc += __shfl_xor(sacc, o, 32);
            if (lane == 0) part[w * 16 + m] = sacc;
        }
    }
    __syncthreads();
    if (t < 16) {
        const float s = part[t] + part[16 + t] + part[32 + t] + part[48 + t];
        rinv[t] = 1.0f / s;
    }
    __syncthreads();

    // ---- Phase 3: quantized = (P @ E) * rinv, wave w owns d in [64w, 64w+64) ----
    v8f qa[4];
#pragma unroll
    for (int j = 0; j < 4; j++)
#pragma unroll
        for (int i = 0; i < 8; i++) qa[j][i] = 0.f;

    const __bf16* ew2l = Ew2 + (size_t)lane * 16;
    for (int kc = 0; kc < 32; kc++) {
        const int kb = kc * 32;
        // A: P rows (16x32 bf16) from LDS
        const __bf16* pp = Pm + hl * PSTR + kb + 8 * g;
        v8bf plo = *(const v8bf*)pp;
        v8bf phi = *(const v8bf*)(pp + 16);
        v16bf A = SHUF16(plo, phi);
#pragma unroll
        for (int j = 0; j < 4; j++) {
            const int dt = 4 * w + j;
            const __bf16* bp = ew2l + (size_t)(dt * 32 + kc) * 512;
            v8bf b0 = *(const v8bf*)bp;
            v8bf b1 = *(const v8bf*)(bp + 8);
            v16bf Bf = SHUF16(b0, b1);
            qa[j] = __builtin_amdgcn_wmma_f32_16x16x32_bf16(
                false, A, false, Bf, (short)0, qa[j], false, false);
        }
    }

    // ---- epilogue: normalize, store, accumulate loss partial ----
    float lacc = 0.f;
#pragma unroll
    for (int j = 0; j < 4; j++) {
#pragma unroll
        for (int r = 0; r < 8; r++) {
            const int m = r + 8 * g;
            const int d = (4 * w + j) * 16 + hl;
            const size_t idx = (size_t)(row0 + m) * Dn + d;
            const float q = qa[j][r] * rinv[m];
            out[idx] = q;
            const float dx = q - x[idx];
            lacc += dx * dx;
        }
    }
    for (int o = 1; o < 32; o <<= 1) lacc += __shfl_xor(lacc, o, 32);
    if (lane == 0) wl[w] = lacc;
    __syncthreads();
    if (t == 0) bpart[blockIdx.x] = wl[0] + wl[1] + wl[2] + wl[3];
}

// ---------------- deterministic loss reduction ----------------
__global__ void svq_loss_reduce(const float* __restrict__ bpart, float* __restrict__ lossout) {
    __shared__ float red[256];
    const int t = threadIdx.x;
    float a = 0.f;
    for (int i = 0; i < NBLK / 256; i++) a += bpart[t + 256 * i];
    red[t] = a;
    __syncthreads();
    for (int o = 128; o > 0; o >>= 1) {
        if (t < o) red[t] += red[t + o];
        __syncthreads();
    }
    if (t == 0) lossout[0] = red[0] * (1.25f / (float)((size_t)Bn * Dn));
}

extern "C" void kernel_launch(void* const* d_in, const int* in_sizes, int n_in,
                              void* d_out, int out_size, void* d_ws, size_t ws_size,
                              hipStream_t stream) {
    (void)in_sizes; (void)n_in; (void)out_size; (void)ws_size;
    const float* x   = (const float*)d_in[0];
    const float* var = (const float*)d_in[1];
    const float* E   = (const float*)d_in[2];
    float*  out   = (float*)d_out;
    float*  ee    = (float*)((char*)d_ws + WS_EE);
    float*  bpart = (float*)((char*)d_ws + WS_BPART);
    __bf16* Ew1   = (__bf16*)((char*)d_ws + WS_EW1);
    __bf16* Ew2   = (__bf16*)((char*)d_ws + WS_EW2);

    svq_prep_ee<<<Kn, 256, 0, stream>>>(E, ee);
    svq_prep_w1<<<64 * 8, 64, 0, stream>>>(E, Ew1);
    svq_prep_w2<<<16 * 32, 64, 0, stream>>>(E, Ew2);
    svq_main<<<NBLK, NTHR, SMEM_SZ, stream>>>(x, var, Ew1, Ew2, ee, out, bpart);
    svq_loss_reduce<<<1, 256, 0, stream>>>(bpart, out + (size_t)Bn * Dn);
}
